// ieHGCN_5583457485247
// MI455X (gfx1250) — matrix operand order repaired and checked by reading, not compiled, not run on hardware
//
#include <hip/hip_runtime.h>
#include <stdint.h>

// ---------------------------------------------------------------------------
// ieHGCN forward (paper branch only; author branch is dead code w.r.t. output)
// bf16 WMMA (v_wmma_f32_16x16x32_bf16) for all [100k,256]x[256,256] GEMMs.
// Each wave register-blocks a 32x64 output strip (8 accumulators, 2 A frags
// sharing 4 B frags): 12 loads per 8 WMMAs. All 8 waves of a block share one
// N-group, whose 32KB B slice is staged into LDS once per block with
// global_load_async_to_lds_b128 (ASYNCcnt) and consumed via ds_load_b128.
// Fused rank-1 attention projections in the GEMM epilogue; atomic scatter for
// the segment_sum SpMM; fused softmax + classifier at the end.
// ---------------------------------------------------------------------------

typedef unsigned short u16b;
typedef unsigned int   u32b;
typedef __attribute__((ext_vector_type(4)))  unsigned int u32x4;
typedef __attribute__((ext_vector_type(4)))  float        f32x4;
typedef __attribute__((ext_vector_type(8)))  float        v8f;
typedef __attribute__((ext_vector_type(16))) __bf16       v16bf;

union Frag16 { u32x4 u[2]; v16bf v; };

__device__ __forceinline__ u16b f32_to_bf16_rne(float f) {
    u32b u = __float_as_uint(f);
    u32b r = u + 0x7FFFu + ((u >> 16) & 1u);   // round-to-nearest-even
    return (u16b)(r >> 16);
}

__device__ __forceinline__ float eluf(float x) {
    return x > 0.f ? x : (expf(x) - 1.f);
}

// --------------------------- f32 -> bf16 cast ------------------------------
__global__ void iehgcn_cast_bf16(const float* __restrict__ in,
                                 u16b* __restrict__ out, int n4) {
    int i = blockIdx.x * blockDim.x + threadIdx.x;
    if (i >= n4) return;
    f32x4 v = *(const f32x4*)(in + (size_t)i * 4);
    u16b* o = out + (size_t)i * 4;
    o[0] = f32_to_bf16_rne(v.x); o[1] = f32_to_bf16_rne(v.y);
    o[2] = f32_to_bf16_rne(v.z); o[3] = f32_to_bf16_rne(v.w);
}

// ------------------- repack W[256,ncols] into B fragments ------------------
// Layout: [ntile][kchunk(8)][lane(32)][dword(8)]; lane g=lane>>4, n=lane&15:
//   dword j packs bf16(W[kb*32+16g+2j, n0+n]) | bf16(W[.. +1, ..])<<16
__global__ void iehgcn_repack_b(const float* __restrict__ W,
                                u32b* __restrict__ Bfrag, int ncols) {
    int idx = blockIdx.x * blockDim.x + threadIdx.x;
    int total = (ncols >> 4) * 8 * 32 * 8;
    if (idx >= total) return;
    int j     = idx & 7;
    int lane  = (idx >> 3) & 31;
    int kb    = (idx >> 8) & 7;
    int ntile = idx >> 11;
    int g = lane >> 4, nl = lane & 15;
    int n  = ntile * 16 + nl;
    int k0 = kb * 32 + 16 * g + 2 * j;
    u32b lo = f32_to_bf16_rne(W[(size_t)k0 * ncols + n]);
    u32b hi = f32_to_bf16_rne(W[(size_t)(k0 + 1) * ncols + n]);
    Bfrag[idx] = lo | (hi << 16);
}

// ------------------------------- init --------------------------------------
__global__ void iehgcn_init(float* __restrict__ conv_acc, float* __restrict__ deg,
                            float* __restrict__ hl, float* __restrict__ hr,
                            float* __restrict__ attrel,
                            const float* __restrict__ wal_b,
                            const float* __restrict__ war_b,
                            int n, int nd) {
    int i = blockIdx.x * blockDim.x + threadIdx.x;
    if (i < nd) conv_acc[i] = 0.f;
    if (i < n) {
        deg[i]    = 0.f;
        hl[i]     = wal_b[0];   // bias of Wal projection
        hr[i]     = war_b[0];   // bias of War projection
        attrel[i] = wal_b[0];   // bias for conv@Wk@Wal projection
    }
}

// ----------------------- generic bf16 WMMA GEMM ----------------------------
// Y[M,256] = A[M,256](bf16) @ W + bias  (W pre-repacked into Bfrag layout).
// Block: 8 waves, one shared N-group (64 cols, B slice = 32KB staged in LDS).
// Wave: one 32-row strip -> 8 accumulators (2 M-strips x 4 N-tiles).
__global__ void iehgcn_wmma_gemm(const u16b* __restrict__ A,
                                 const u32b* __restrict__ Bfrag,
                                 const float* __restrict__ bias,
                                 float* __restrict__ Yf,
                                 u16b* __restrict__ Ybf,
                                 const float* __restrict__ projv,
                                 float* __restrict__ projout,
                                 int M) {
    __shared__ __align__(16) u32b smemB[8192];          // 32KB: 4 ntiles x 8 kb x 1KB
    const int tid  = threadIdx.x;
    const int lane = tid & 31;
    const int wv   = tid >> 5;
    const int ng     = blockIdx.x & 3;                  // N-group (cols ng*64..+63)
    const int mblock = blockIdx.x >> 2;

    // --- async-stage this N-group's B slice (contiguous 32KB) into LDS ---
    {
        const u32b* gsrc = Bfrag + (size_t)ng * 8192 + (size_t)tid * 4;
        unsigned    loff = (unsigned)(size_t)(&smemB[0]) + (unsigned)tid * 16;
#pragma unroll
        for (int i = 0; i < 8; ++i) {
            asm volatile("global_load_async_to_lds_b128 %0, %1, off"
                         :: "v"(loff + i * 4096), "v"(gsrc + i * 1024)
                         : "memory");
        }
        asm volatile("s_wait_asynccnt 0x0" ::: "memory");
    }
    __syncthreads();

    const int strip = mblock * 8 + wv;                  // 32-row strip
    if (strip >= (M >> 5)) return;                      // wave-uniform guard

    const int g  = lane >> 4;                           // half-wave group
    const int nl = lane & 15;                           // lane within group
    const u16b* Arow0 = A + (size_t)(strip * 32 + nl) * 256;
    const u16b* Arow1 = Arow0 + 16 * 256;
    const u32b* lB    = &smemB[lane * 8];

    v8f zero = {};
    v8f c0[4], c1[4];
#pragma unroll
    for (int t = 0; t < 4; ++t) { c0[t] = zero; c1[t] = zero; }

#pragma unroll
    for (int kb = 0; kb < 8; ++kb) {
        Frag16 a0, a1;
        // A 16x32 bf16 fragment: VGPR0-3 <- K = kb*32+16g+{0..7},
        //                        VGPR4-7 <- K = kb*32+16g+8+{0..7}
        a0.u[0] = *(const u32x4*)(Arow0 + kb * 32 + 16 * g);
        a0.u[1] = *(const u32x4*)(Arow0 + kb * 32 + 16 * g + 8);
        a1.u[0] = *(const u32x4*)(Arow1 + kb * 32 + 16 * g);
        a1.u[1] = *(const u32x4*)(Arow1 + kb * 32 + 16 * g + 8);
#pragma unroll
        for (int t = 0; t < 4; ++t) {
            Frag16 b;
            const u32x4* bp = (const u32x4*)(lB + t * 2048 + kb * 256);
            b.u[0] = bp[0];
            b.u[1] = bp[1];
            c0[t] = __builtin_amdgcn_wmma_f32_16x16x32_bf16(
                        false, a0.v, false, b.v, (short)0, c0[t], false, false);
            c1[t] = __builtin_amdgcn_wmma_f32_16x16x32_bf16(
                        false, a1.v, false, b.v, (short)0, c1[t], false, false);
        }
    }

    // C/D layout: VGPR r -> M = strip*32 (+16) + 8g + r, N = ng*64 + 16t + nl
    const int n0 = ng * 64 + nl;
    float bs[4] = {0.f, 0.f, 0.f, 0.f};
    if (bias) {
#pragma unroll
        for (int t = 0; t < 4; ++t) bs[t] = bias[n0 + 16 * t];
    }

    if (Yf || Ybf) {
#pragma unroll
        for (int r = 0; r < 8; ++r) {
            int m0 = strip * 32 + g * 8 + r;
            size_t ro0 = (size_t)m0 * 256 + n0;
            size_t ro1 = ro0 + 16 * 256;
#pragma unroll
            for (int t = 0; t < 4; ++t) {
                float v0 = c0[t][r] + bs[t];
                float v1 = c1[t][r] + bs[t];
                if (Yf)  { Yf[ro0 + 16 * t] = v0;  Yf[ro1 + 16 * t] = v1; }
                if (Ybf) { Ybf[ro0 + 16 * t] = f32_to_bf16_rne(v0);
                           Ybf[ro1 + 16 * t] = f32_to_bf16_rne(v1); }
            }
        }
    }
    if (projv) {
        float p[4];
#pragma unroll
        for (int t = 0; t < 4; ++t) p[t] = projv[n0 + 16 * t];
#pragma unroll
        for (int r = 0; r < 8; ++r) {
            // combine the 4 resident N-tiles in registers first
            float v0 = 0.f, v1 = 0.f;
#pragma unroll
            for (int t = 0; t < 4; ++t) {
                v0 += (c0[t][r] + bs[t]) * p[t];
                v1 += (c1[t][r] + bs[t]) * p[t];
            }
            v0 += __shfl_xor(v0, 1, 32); v1 += __shfl_xor(v1, 1, 32);
            v0 += __shfl_xor(v0, 2, 32); v1 += __shfl_xor(v1, 2, 32);
            v0 += __shfl_xor(v0, 4, 32); v1 += __shfl_xor(v1, 4, 32);
            v0 += __shfl_xor(v0, 8, 32); v1 += __shfl_xor(v1, 8, 32);
            if (nl == 0) {
                int m = strip * 32 + g * 8 + r;
                atomicAdd(projout + m,      v0);
                atomicAdd(projout + m + 16, v1);
            }
        }
    }
}

// -------------------- edge scatter (segment_sum) ---------------------------
// 64 threads per edge, float4 each: conv_acc[dst] += msg[src]; deg[dst] += 1.
__global__ void iehgcn_scatter(const float* __restrict__ msg,
                               const int* __restrict__ esrc,
                               const int* __restrict__ edst,
                               float* __restrict__ conv_acc,
                               float* __restrict__ deg, int E) {
    long idx = (long)blockIdx.x * blockDim.x + threadIdx.x;
    int e = (int)(idx >> 6);
    if (e >= E) return;
    int q = (int)(idx & 63);
    int s = esrc[e], d = edst[e];
    f32x4 v = *(const f32x4*)(msg + (size_t)s * 256 + q * 4);
    float* dp = conv_acc + (size_t)d * 256 + q * 4;
    atomicAdd(dp + 0, v.x);
    atomicAdd(dp + 1, v.y);
    atomicAdd(dp + 2, v.z);
    atomicAdd(dp + 3, v.w);
    if (q == 0) atomicAdd(deg + d, 1.f);
}

// -------------------- conv = acc / max(deg,1) + bias -----------------------
__global__ void iehgcn_finalize_conv(float* __restrict__ conv,
                                     const float* __restrict__ deg,
                                     const float* __restrict__ convb,
                                     u16b* __restrict__ convbf, int nd) {
    int i = blockIdx.x * blockDim.x + threadIdx.x;
    if (i >= nd) return;
    int m = i >> 8, cix = i & 255;
    float dg = deg[m];
    if (dg < 1.f) dg = 1.f;
    float v = conv[i] / dg + convb[cix];
    conv[i]   = v;
    convbf[i] = f32_to_bf16_rne(v);
}

// ----------------- elu + 2-way softmax attention weights -------------------
__global__ void iehgcn_attention(float* __restrict__ hl, float* __restrict__ hr,
                                 const float* __restrict__ attrel, int n) {
    int m = blockIdx.x * blockDim.x + threadIdx.x;
    if (m >= n) return;
    float r  = hr[m];
    float e0 = eluf(hl[m] + r);
    float e1 = eluf(attrel[m] + r);
    float mx = fmaxf(e0, e1);
    float a0 = expf(e0 - mx), a1 = expf(e1 - mx);
    float s  = a0 + a1;
    hl[m] = a0 / s;     // att[self]
    hr[m] = a1 / s;     // att[relation]
}

// ----- out = (a0*dst + a1*conv) @ cls_w + cls_b ; one wave per row ---------
__global__ void iehgcn_classifier(const float* __restrict__ dstp,
                                  const float* __restrict__ conv,
                                  const float* __restrict__ a0v,
                                  const float* __restrict__ a1v,
                                  const float* __restrict__ clsw,  // [256,16]
                                  const float* __restrict__ clsb,  // [16]
                                  float* __restrict__ out, int N) {
    int row  = blockIdx.x * (blockDim.x >> 5) + (threadIdx.x >> 5);
    int lane = threadIdx.x & 31;
    if (row >= N) return;
    float w0 = a0v[row], w1 = a1v[row];
    float acc[16];
#pragma unroll
    for (int c = 0; c < 16; ++c) acc[c] = 0.f;
    for (int n = lane; n < 256; n += 32) {
        float r = w0 * dstp[(size_t)row * 256 + n] + w1 * conv[(size_t)row * 256 + n];
        const f32x4* wp = (const f32x4*)(clsw + (size_t)n * 16);
#pragma unroll
        for (int q = 0; q < 4; ++q) {
            f32x4 cw = wp[q];
            acc[q * 4 + 0] += r * cw.x;
            acc[q * 4 + 1] += r * cw.y;
            acc[q * 4 + 2] += r * cw.z;
            acc[q * 4 + 3] += r * cw.w;
        }
    }
#pragma unroll
    for (int c = 0; c < 16; ++c) {
        acc[c] += __shfl_xor(acc[c], 16, 32);
        acc[c] += __shfl_xor(acc[c],  8, 32);
        acc[c] += __shfl_xor(acc[c],  4, 32);
        acc[c] += __shfl_xor(acc[c],  2, 32);
        acc[c] += __shfl_xor(acc[c],  1, 32);
    }
    if (lane == 0) {
#pragma unroll
        for (int c = 0; c < 16; ++c)
            out[(size_t)row * 16 + c] = acc[c] + clsb[c];
    }
}

// ---------------------------------------------------------------------------
extern "C" void kernel_launch(void* const* d_in, const int* in_sizes, int n_in,
                              void* d_out, int out_size, void* d_ws, size_t ws_size,
                              hipStream_t stream) {
    // setup_inputs() dict order
    const float* x_paper  = (const float*)d_in[0];
    const float* x_author = (const float*)d_in[1];
    const int*   e_w_src  = (const int*)d_in[2];
    const int*   e_w_dst  = (const int*)d_in[3];
    // d_in[4], d_in[5]: written_by edges (dead code for the output)
    const float* Wself_p_w = (const float*)d_in[6];
    const float* Wself_p_b = (const float*)d_in[7];
    const float* Wq_p_w    = (const float*)d_in[8];
    const float* Wq_p_b    = (const float*)d_in[9];
    const float* Wk_p_w    = (const float*)d_in[10];
    const float* Wk_p_b    = (const float*)d_in[11];
    const float* Wal_p_w   = (const float*)d_in[12];
    const float* Wal_p_b   = (const float*)d_in[13];
    const float* War_p_w   = (const float*)d_in[14];
    const float* War_p_b   = (const float*)d_in[15];
    // 16..25: author-type weights (dead)
    const float* conv_w = (const float*)d_in[26];
    const float* conv_b = (const float*)d_in[27];
    // 28,29: conv_wb (dead)
    const float* cls_w  = (const float*)d_in[30];
    const float* cls_b  = (const float*)d_in[31];
    float* out = (float*)d_out;

    const int D  = 256;
    const int N  = in_sizes[0] / D;     // 100000 (multiple of 32)
    const int NA = in_sizes[1] / D;
    const int E  = in_sizes[2];
    const int ND = N * D;
    const int NAD = NA * D;

    // ---- workspace carve-out ----
    char* base = (char*)d_ws;
    size_t off = 0;
    auto carve = [&](size_t bytes) -> void* {
        void* p = base + off;
        off += (bytes + 255) & ~(size_t)255;
        return p;
    };
    u16b*  xp_bf    = (u16b*)carve((size_t)ND * 2);
    u16b*  xa_bf    = (u16b*)carve((size_t)NAD * 2);
    float* dstp_f   = (float*)carve((size_t)ND * 4);
    u16b*  dstp_bf  = (u16b*)carve((size_t)ND * 2);
    float* msg_f    = (float*)carve((size_t)NAD * 4);
    float* conv_f   = (float*)carve((size_t)ND * 4);   // acc, then normalized in place
    u16b*  conv_bf  = (u16b*)carve((size_t)ND * 2);
    float* deg      = (float*)carve((size_t)N * 4);
    float* hl       = (float*)carve((size_t)N * 4);
    float* hr       = (float*)carve((size_t)N * 4);
    float* attrel   = (float*)carve((size_t)N * 4);
    u32b*  fragSelf = (u32b*)carve(32768 * 4);
    u32b*  fragK    = (u32b*)carve(32768 * 4);
    u32b*  fragQ    = (u32b*)carve(32768 * 4);
    u32b*  fragConv = (u32b*)carve(32768 * 4);

    auto cdiv = [](int a, int b) { return (a + b - 1) / b; };

    // 1) cast features to bf16
    iehgcn_cast_bf16<<<cdiv(ND / 4, 256), 256, 0, stream>>>(x_paper, xp_bf, ND / 4);
    iehgcn_cast_bf16<<<cdiv(NAD / 4, 256), 256, 0, stream>>>(x_author, xa_bf, NAD / 4);

    // 2) repack weights into WMMA B-fragment layout
    iehgcn_repack_b<<<cdiv(32768, 256), 256, 0, stream>>>(Wself_p_w, fragSelf, 256);
    iehgcn_repack_b<<<cdiv(32768, 256), 256, 0, stream>>>(Wk_p_w,    fragK,    256);
    iehgcn_repack_b<<<cdiv(32768, 256), 256, 0, stream>>>(Wq_p_w,    fragQ,    256);
    iehgcn_repack_b<<<cdiv(32768, 256), 256, 0, stream>>>(conv_w,    fragConv, 256);

    // 3) init accumulators / bias-seeded projection outputs
    iehgcn_init<<<cdiv(ND, 256), 256, 0, stream>>>(conv_f, deg, hl, hr, attrel,
                                                   Wal_p_b, War_p_b, N, ND);

    auto gemm = [&](const u16b* A, const u32b* B, const float* bias, float* Yf,
                    u16b* Ybf, const float* pv, float* po, int M) {
        int nstrips = M / 32;                  // 32-row strip per wave
        int blocks  = cdiv(nstrips, 8) * 4;    // 8 waves/block x 4 N-groups
        iehgcn_wmma_gemm<<<blocks, 256, 0, stream>>>(A, B, bias, Yf, Ybf, pv, po, M);
    };

    // 4) dst_p = x_paper @ Wself + b   (f32 + bf16 copies)
    gemm(xp_bf, fragSelf, Wself_p_b, dstp_f, dstp_bf, nullptr, nullptr, N);
    // 5) hl += (dst_p @ Wk + bk) . Wal   (fused projection)
    gemm(dstp_bf, fragK, Wk_p_b, nullptr, nullptr, Wal_p_w, hl, N);
    // 6) hr += (dst_p @ Wq + bq) . War
    gemm(dstp_bf, fragQ, Wq_p_b, nullptr, nullptr, War_p_w, hr, N);
    // 7) msg = x_author @ conv_w   (bias added after degree-normalization)
    gemm(xa_bf, fragConv, nullptr, msg_f, nullptr, nullptr, nullptr, NA);

    // 8) conv_acc[dst] += msg[src]; deg[dst] += 1
    {
        long threads = (long)E * 64;
        iehgcn_scatter<<<(int)((threads + 255) / 256), 256, 0, stream>>>(
            msg_f, e_w_src, e_w_dst, conv_f, deg, E);
    }
    // 9) conv = acc / max(deg,1) + conv_b
    iehgcn_finalize_conv<<<cdiv(ND, 256), 256, 0, stream>>>(conv_f, deg, conv_b,
                                                            conv_bf, ND);
    // 10) attrel += (conv @ Wk + bk) . Wal
    gemm(conv_bf, fragK, Wk_p_b, nullptr, nullptr, Wal_p_w, attrel, N);

    // 11) softmax([elu(hl+hr), elu(attrel+hr)]) -> (a0, a1) in (hl, hr)
    iehgcn_attention<<<cdiv(N, 256), 256, 0, stream>>>(hl, hr, attrel, N);

    // 12) out = (a0*dst_p + a1*conv) @ cls_w + cls_b
    iehgcn_classifier<<<cdiv(N, 8), 256, 0, stream>>>(dstp_f, conv_f, hl, hr,
                                                      cls_w, cls_b, out, N);
}
